// BidirectionalCrossModalCausalAttention_87222195847285
// MI455X (gfx1250) — matrix operand order repaired and checked by reading, not compiled, not run on hardware
//
#include <hip/hip_runtime.h>
#include <hip/hip_bf16.h>

typedef __attribute__((ext_vector_type(16))) __bf16 v16bf;
typedef __attribute__((ext_vector_type(8)))  float  v8f;

#define B_   64
#define S_   4
#define C_   1280
#define HID_ 256
#define HW_  1024
#define EPS_ 1e-8f

__device__ __forceinline__ unsigned pk2bf(float a, float b) {
  union { __bf16 h[2]; unsigned u; } r;
  r.h[0] = (__bf16)a; r.h[1] = (__bf16)b;
  return r.u;
}

// ---------------- kernel 0: normalized sensor query ----------------
__global__ void qn_kernel(const float* __restrict__ sensor, const float* __restrict__ Wq,
                          const float* __restrict__ bq, float* __restrict__ qn_ws) {
  __shared__ float red[256];
  int b = blockIdx.x, t = threadIdx.x;
  float s0 = sensor[b*4+0], s1 = sensor[b*4+1], s2 = sensor[b*4+2], s3 = sensor[b*4+3];
  float q = bq[t] + s0*Wq[0*HID_+t] + s1*Wq[1*HID_+t] + s2*Wq[2*HID_+t] + s3*Wq[3*HID_+t];
  red[t] = q*q;
  __syncthreads();
  for (int off = 128; off > 0; off >>= 1) {
    if (t < off) red[t] += red[t+off];
    __syncthreads();
  }
  float norm = sqrtf(red[0]);
  qn_ws[b*HID_ + t] = q / fmaxf(norm, EPS_);
}

// ---------------- kernel 1: k-GEMM + cosine scores (WMMA bf16, pipelined) ----------------
// Block: 256 threads = 8 waves.  Block tile: M=256 (full HID) x N=64 pixels.
// Wave w: mw=w&3 -> 64 M rows, nw=w>>2 -> 32 N cols.  K loop: 1280 in steps of 32.
// Software pipeline: next K-step's fp32 Wk/x values held in registers while
// current step's WMMAs run; bf16 conversion happens at store time (packed b32).
__global__ __launch_bounds__(256)
void scores_kernel(const float* __restrict__ x, const float* __restrict__ Wk,
                   const float* __restrict__ bk, const float* __restrict__ qn_ws,
                   float* __restrict__ scores_ws) {
  __shared__ unsigned As_u[256*20];   // A[m][kk] as bf16 pairs, row stride 20 uints (40 bf16)
  __shared__ unsigned Bs_u[64*20];    // B[n][kk] as bf16 pairs, row stride 20 uints
  __shared__ float  red_kk[512];
  __shared__ float  red_qk[512];
  __shared__ float  qn_l[256];
  __shared__ float  bk_l[256];

  const __bf16* As = (const __bf16*)As_u;
  const __bf16* Bs = (const __bf16*)Bs_u;

  const int b    = blockIdx.y;
  const int pix0 = blockIdx.x * 64;
  const int t    = threadIdx.x;
  const int lane = t & 31;
  const int w    = t >> 5;
  const int mw   = w & 3;
  const int nw   = w >> 2;

  qn_l[t] = qn_ws[b*HID_ + t];
  bk_l[t] = bk[t];

  const float* xb     = x + (size_t)b * C_ * HW_ + pix0;
  const float* WkRow  = Wk + (size_t)t * C_;
  const int kkB = (t & 15) * 2;      // this thread's pair of channels in the B tile
  const int nB  = (t >> 4) * 4;      // this thread's 4 pixels in the B tile

  // fp32 staging registers for the *next* K-step (pipelined)
  float aReg[32];
  float bReg[8];

  // prologue: fetch K-step 0
#pragma unroll
  for (int j = 0; j < 32; j++) aReg[j] = WkRow[j];
  {
    const float* p = xb + (size_t)kkB * HW_ + nB;
#pragma unroll
    for (int j = 0; j < 4; j++) { bReg[j] = p[j]; bReg[4+j] = p[HW_ + j]; }
  }

  v8f acc[4][2];
  const v8f vzero = {0.f,0.f,0.f,0.f,0.f,0.f,0.f,0.f};
#pragma unroll
  for (int mi = 0; mi < 4; mi++)
#pragma unroll
    for (int ni = 0; ni < 2; ni++)
      acc[mi][ni] = vzero;

  for (int k0 = 0; k0 < C_; k0 += 32) {
    __syncthreads();   // previous step's fragment reads complete
    // Stage A tile: pack 32 floats -> 16 contiguous uints (merges to ds_store_b128 x4)
    {
      unsigned* au = As_u + t * 20;
#pragma unroll
      for (int j = 0; j < 16; j++) au[j] = pk2bf(aReg[2*j], aReg[2*j+1]);
    }
    // Stage B tile: 4 pixels x 1 packed channel-pair -> 4x ds_store_b32
    {
      const int cu = kkB >> 1;
#pragma unroll
      for (int j = 0; j < 4; j++) Bs_u[(nB + j)*20 + cu] = pk2bf(bReg[j], bReg[4+j]);
    }
    __syncthreads();

    // Issue next K-step's global loads now; WMMAs below hide the latency.
    if (k0 + 32 < C_) {
      const float* wr = WkRow + k0 + 32;
#pragma unroll
      for (int j = 0; j < 32; j++) aReg[j] = wr[j];
      const float* p = xb + (size_t)(k0 + 32 + kkB) * HW_ + nB;
#pragma unroll
      for (int j = 0; j < 4; j++) { bReg[j] = p[j]; bReg[4+j] = p[HW_ + j]; }
    }

    // B fragments: lane l holds N=l%16, K = (l<16?0:16)+i for i=0..15
    v16bf bf[2];
    const int kbB = (lane < 16) ? 0 : 16;
#pragma unroll
    for (int ni = 0; ni < 2; ni++) {
      const __bf16* br = Bs + (nw*32 + ni*16 + (lane & 15)) * 40 + kbB;
#pragma unroll
      for (int i = 0; i < 16; i++) bf[ni][i] = br[i];
    }
    // A fragments: lane l holds M=l%16; K = base..base+15, base=(l<16?0:16)
    const int kbA = (lane < 16) ? 0 : 16;
#pragma unroll
    for (int mi = 0; mi < 4; mi++) {
      const __bf16* ar = As + (mw*64 + mi*16 + (lane & 15)) * 40;
      v16bf af;
#pragma unroll
      for (int i = 0; i < 8; i++) af[i]     = ar[kbA + i];
#pragma unroll
      for (int i = 0; i < 8; i++) af[8 + i] = ar[kbA + 8 + i];
#pragma unroll
      for (int ni = 0; ni < 2; ni++)
        acc[mi][ni] = __builtin_amdgcn_wmma_f32_16x16x32_bf16(
            false, af, false, bf[ni], (short)0, acc[mi][ni], false, false);
    }
  }

  // Epilogue: per-pixel ||k||^2 and qn.k partials. C layout: VGPR r -> M = r (lanes<16) or 8+r.
  float kkp[2] = {0.f, 0.f}, qkp[2] = {0.f, 0.f};
#pragma unroll
  for (int mi = 0; mi < 4; mi++) {
#pragma unroll
    for (int r = 0; r < 8; r++) {
      int m = mw*64 + mi*16 + ((lane < 16) ? r : 8 + r);
      float bkm = bk_l[m], qnm = qn_l[m];
#pragma unroll
      for (int ni = 0; ni < 2; ni++) {
        float val = acc[mi][ni][r] + bkm;
        kkp[ni] += val * val;
        qkp[ni] += qnm * val;
      }
    }
  }
#pragma unroll
  for (int ni = 0; ni < 2; ni++) {
    int slot = w*64 + ni*16 + (lane & 15) + ((lane >> 4) << 5);
    red_kk[slot] = kkp[ni];
    red_qk[slot] = qkp[ni];
  }
  __syncthreads();
  if (t < 64) {
    int n   = t;
    int nwn = n >> 5;
    int loc = n & 31;
    int base0 = (loc >> 4) * 16 + (loc & 15);
    float skk = 0.f, sqk = 0.f;
#pragma unroll
    for (int mwi = 0; mwi < 4; mwi++) {
      int wb = (nwn*4 + mwi)*64 + base0;
      skk += red_kk[wb] + red_kk[wb + 32];
      sqk += red_qk[wb] + red_qk[wb + 32];
    }
    scores_ws[b*HW_ + pix0 + n] = sqk / fmaxf(sqrtf(skk), EPS_);
  }
}

// ---------------- kernel 2: softmax over HW per batch -> attn ----------------
__global__ void softmax_kernel(const float* __restrict__ scores_ws, float* __restrict__ attn_out) {
  __shared__ float red[256];
  int b = blockIdx.x, t = threadIdx.x;
  const float* s = scores_ws + b*HW_;
  float v[4];
  float mx = -1e30f;
#pragma unroll
  for (int j = 0; j < 4; j++) { v[j] = s[t + j*256]; mx = fmaxf(mx, v[j]); }
  red[t] = mx; __syncthreads();
  for (int off = 128; off > 0; off >>= 1) {
    if (t < off) red[t] = fmaxf(red[t], red[t+off]);
    __syncthreads();
  }
  mx = red[0];
  __syncthreads();
  float sum = 0.f;
#pragma unroll
  for (int j = 0; j < 4; j++) { v[j] = expf(v[j] - mx); sum += v[j]; }
  red[t] = sum; __syncthreads();
  for (int off = 128; off > 0; off >>= 1) {
    if (t < off) red[t] += red[t+off];
    __syncthreads();
  }
  float inv = 1.f / red[0];
#pragma unroll
  for (int j = 0; j < 4; j++) attn_out[b*HW_ + t + j*256] = v[j] * inv;
}

// ---------------- kernel 3: one pass over x -> gap (mean) and xw = sum(attn*x) ----------------
__global__ void reduce_x_kernel(const float* __restrict__ x, const float* __restrict__ attn,
                                float* __restrict__ gap_ws, float* __restrict__ xw_ws) {
  __shared__ float attn_l[1024];
  __shared__ float r1[256], r2[256];
  int b = blockIdx.y, c0 = blockIdx.x * 16, t = threadIdx.x;
  const float* ab = attn + b*HW_;
#pragma unroll
  for (int j = 0; j < 4; j++) attn_l[t + j*256] = ab[t + j*256];
  __syncthreads();
  int ci = c0 + (t >> 4);
  int s  = t & 15;
  const float* xr = x + (size_t)b*C_*HW_ + (size_t)ci*HW_;
  float ssum = 0.f, wsum = 0.f;
  for (int hw = s; hw < HW_; hw += 16) {
    float v = xr[hw];
    ssum += v;
    wsum += attn_l[hw] * v;
  }
  r1[t] = ssum; r2[t] = wsum; __syncthreads();
  for (int off = 8; off > 0; off >>= 1) {
    if (s < off) { r1[t] += r1[t+off]; r2[t] += r2[t+off]; }
    __syncthreads();
  }
  if (s == 0) {
    gap_ws[b*C_ + ci] = r1[t] * (1.f/1024.f);
    xw_ws[b*C_ + ci]  = r2[t];
  }
}

// ---------------- kernel 4: visual_vector = Wp.(Wv.xw+bv)/1024 + bp ----------------
__global__ void vv_kernel(const float* __restrict__ xw_ws, const float* __restrict__ Wv,
                          const float* __restrict__ bv, const float* __restrict__ Wp,
                          const float* __restrict__ bp, float* __restrict__ vv_out) {
  __shared__ float xw_l[1280];
  __shared__ float t1_l[256];
  int b = blockIdx.x, t = threadIdx.x;
#pragma unroll
  for (int j = 0; j < 5; j++) xw_l[t + j*256] = xw_ws[b*C_ + t + j*256];
  __syncthreads();
  {
    const float* wr = Wv + (size_t)t * C_;
    float acc = bv[t];
    for (int c = 0; c < C_; c++) acc += wr[c] * xw_l[c];
    t1_l[t] = acc * (1.f/1024.f);   // pooled_mv
  }
  __syncthreads();
#pragma unroll
  for (int j = 0; j < 5; j++) {
    int c = t + j*256;
    const float* wr = Wp + (size_t)c * HID_;
    float acc = bp[c];
    for (int k = 0; k < HID_; k++) acc += wr[k] * t1_l[k];
    vv_out[b*C_ + c] = acc;
  }
}

// ---------------- kernel 5: gap -> MLP -> softmax gate -> recalibrated sensor ----------------
__global__ void mlp_kernel(const float* __restrict__ gap_ws, const float* __restrict__ W1,
                           const float* __restrict__ b1, const float* __restrict__ W2,
                           const float* __restrict__ b2, const float* __restrict__ sensor,
                           float* __restrict__ recal_out, float* __restrict__ sw_out) {
  __shared__ float gap_l[1280];
  __shared__ float h_l[256];
  __shared__ float logits[4];
  int b = blockIdx.x, t = threadIdx.x;
#pragma unroll
  for (int j = 0; j < 5; j++) gap_l[t + j*256] = gap_ws[b*C_ + t + j*256];
  __syncthreads();
  {
    float acc = b1[t];
    for (int c = 0; c < C_; c++) acc += gap_l[c] * W1[c*HID_ + t];
    h_l[t] = fmaxf(acc, 0.f);
  }
  __syncthreads();
  if (t < 4) {
    float acc = b2[t];
    for (int k = 0; k < HID_; k++) acc += h_l[k] * W2[k*4 + t];
    logits[t] = acc;
  }
  __syncthreads();
  if (t < 4) {
    float mx = fmaxf(fmaxf(logits[0], logits[1]), fmaxf(logits[2], logits[3]));
    float e0 = expf(logits[0]-mx), e1 = expf(logits[1]-mx),
          e2 = expf(logits[2]-mx), e3 = expf(logits[3]-mx);
    float wgt = expf(logits[t]-mx) / (e0+e1+e2+e3);
    sw_out[b*4 + t]    = wgt;
    recal_out[b*4 + t] = sensor[b*4 + t] * wgt;
  }
}

extern "C" void kernel_launch(void* const* d_in, const int* in_sizes, int n_in,
                              void* d_out, int out_size, void* d_ws, size_t ws_size,
                              hipStream_t stream) {
  const float* sensor = (const float*)d_in[0];
  const float* x      = (const float*)d_in[1];
  const float* Wq = (const float*)d_in[2];
  const float* bq = (const float*)d_in[3];
  const float* Wk = (const float*)d_in[4];
  const float* bk = (const float*)d_in[5];
  const float* Wv = (const float*)d_in[6];
  const float* bv = (const float*)d_in[7];
  const float* Wp = (const float*)d_in[8];
  const float* bp = (const float*)d_in[9];
  const float* W1 = (const float*)d_in[10];
  const float* b1 = (const float*)d_in[11];
  const float* W2 = (const float*)d_in[12];
  const float* b2 = (const float*)d_in[13];

  float* out      = (float*)d_out;
  float* vv_out   = out;                          // B*C   = 81920
  float* recal    = out + B_*C_;                  // B*S   = 256
  float* attn_out = out + B_*C_ + B_*S_;          // B*HW  = 65536
  float* sw_out   = out + B_*C_ + B_*S_ + B_*HW_; // B*S   = 256

  float* ws        = (float*)d_ws;
  float* qn_ws     = ws;                          // B*HID = 16384
  float* scores_ws = qn_ws + B_*HID_;             // B*HW  = 65536
  float* gap_ws    = scores_ws + B_*HW_;          // B*C   = 81920
  float* xw_ws     = gap_ws + B_*C_;              // B*C   = 81920

  qn_kernel      <<<B_, 256, 0, stream>>>(sensor, Wq, bq, qn_ws);
  scores_kernel  <<<dim3(HW_/64, B_), 256, 0, stream>>>(x, Wk, bk, qn_ws, scores_ws);
  softmax_kernel <<<B_, 256, 0, stream>>>(scores_ws, attn_out);
  reduce_x_kernel<<<dim3(C_/16, B_), 256, 0, stream>>>(x, attn_out, gap_ws, xw_ws);
  vv_kernel      <<<B_, 256, 0, stream>>>(xw_ws, Wv, bv, Wp, bp, vv_out);
  mlp_kernel     <<<B_, 256, 0, stream>>>(gap_ws, W1, b1, W2, b2, sensor, recal, sw_out);
}
